// EfficientViT_26061861552488
// MI455X (gfx1250) — matrix operand order repaired and checked by reading, not compiled, not run on hardware
//
#include <hip/hip_runtime.h>
#include <hip/hip_bf16.h>
#include <type_traits>

// ---------------------------------------------------------------------------
// EfficientViT linear-attention block for MI455X (gfx1250, wave32, WMMA).
// Heavy GEMMs on v_wmma_f32_16x16x32_f16 (f16 in, f32 accumulate).
// Tile: BM=64 x BN=128 x BK=32; 8 waves, each computes 32x32 (4 WMMAs/K-step,
// 2x fragment reuse). f16 operands staged with async global->LDS copies
// (double-buffered); f32 operands staged via packed register conversion.
// ---------------------------------------------------------------------------

typedef __attribute__((ext_vector_type(16))) _Float16 v16h;
typedef __attribute__((ext_vector_type(8)))  _Float16 v8h;
typedef __attribute__((ext_vector_type(8)))  float    v8f;
typedef int v4i_vs __attribute__((vector_size(16)));   // matches builtin pointee

#define BATCH   16
#define DIM     512
#define DHALF   256
#define ATTN    768      // 3*DHALF (qkv channels)
#define INNER   512      // 2*DHALF
#define NPIX    1024     // 32*32
#define EPSV    1e-5f

#define BPAD    40       // padded B-tile row stride in halfs (80B, 16B-aligned)

#if __has_builtin(__builtin_amdgcn_global_load_async_to_lds_b128)
#define HAVE_ASYNC_LDS 1
#else
#define HAVE_ASYNC_LDS 0
#endif

__device__ __forceinline__ void copy16_g2l(const _Float16* __restrict__ g,
                                           _Float16* __restrict__ l) {
#if HAVE_ASYNC_LDS
    __builtin_amdgcn_global_load_async_to_lds_b128(
        (__attribute__((address_space(1))) v4i_vs*)g,
        (__attribute__((address_space(3))) v4i_vs*)l, 0, 0);
#else
    *(v8h*)l = *(const v8h*)g;
#endif
}

__device__ __forceinline__ void wait_async_all() {
#if HAVE_ASYNC_LDS
#if __has_builtin(__builtin_amdgcn_s_wait_asynccnt)
    __builtin_amdgcn_s_wait_asynccnt(0);
#else
    asm volatile("s_wait_asynccnt 0x0" ::: "memory");
#endif
#endif
}

__device__ __forceinline__ v8h pack8_strided(const float* __restrict__ s, long stride) {
    v8h r;
    #pragma unroll
    for (int i = 0; i < 8; ++i) r[i] = (_Float16)s[i * stride];
    return r;
}

__device__ __forceinline__ v16h join16(v8h lo, v8h hi) {
    v16h r;
    #pragma unroll
    for (int i = 0; i < 8; ++i) { r[i] = lo[i]; r[i + 8] = hi[i]; }
    return r;
}

// ---------------------------------------------------------------------------
// Tiled WMMA GEMM:  C[b] = epilogue( A[b] (MxK,f16) * B[b] (KxN) )
// BT=true  -> B stored as [N][K] row-major f16 (async-staged copy)
// BT=false -> B stored as [K][N] row-major f32 (register-transposed staging)
// MODE 0: f32 out = acc*scale[m] + bias[m]                        (conv+BN)
// MODE 1: f16 out = acc*alpha                                     (vk * 1/N)
// MODE 2: f32 out = (acc/z[b,n])*scale[m] + bias[m] + vres[b,m,n] (attn)
// ---------------------------------------------------------------------------
template <int MODE, bool BT>
__global__ __launch_bounds__(256) void gemm_wmma_kernel(
    const _Float16* __restrict__ A, const void* __restrict__ Bv,
    void* __restrict__ Cv,
    int M, int N, int K,
    long strideA, long strideB, long strideC,
    const float* __restrict__ scale, const float* __restrict__ bias,
    float alpha,
    const float* __restrict__ z,        // MODE 2: [BATCH, N]
    const _Float16* __restrict__ vres)  // MODE 2: f16, same layout as C
{
    using OutT = typename std::conditional<MODE == 1, _Float16, float>::type;
    const int bz = blockIdx.z;
    const _Float16* Ab = A + (long)bz * strideA;
    const _Float16* Bh = BT ? ((const _Float16*)Bv + (long)bz * strideB) : nullptr;
    const float*    Bf = BT ? nullptr : ((const float*)Bv + (long)bz * strideB);
    OutT*           Cb = (OutT*)Cv + (long)bz * strideC;

    const int tid  = threadIdx.x;
    const int lane = tid & 31;
    const int wave = tid >> 5;
    const int M0   = blockIdx.y * 64;
    const int N0   = blockIdx.x * 128;

    const int moff = (wave & 1) * 32;          // 2 waves over M
    const int noff = (wave >> 1) * 32;         // 4 waves over N
    const int r    = lane & 15;
    const int half = lane >> 4;

    __shared__ alignas(16) _Float16 a_s[2][64][32];      // [buf][m][k]
    __shared__ alignas(16) _Float16 b_s[2][128][BPAD];   // [buf][n][k]

    // stage A tile (64x32 f16 = 256 x 16B): one async b128 per thread
    auto stageA = [&](int buf, int k0) {
        const int row = tid >> 2;
        const int seg = (tid & 3) * 8;
        copy16_g2l(Ab + (long)(M0 + row) * K + k0 + seg, &a_s[buf][row][seg]);
    };
    // stage B tile, BT (128x32 f16 = 512 x 16B): two async b128 per thread
    auto stageB_half = [&](int buf, int k0) {
        #pragma unroll
        for (int t = 0; t < 2; ++t) {
            const int idx = tid + t * 256;
            const int row = idx >> 2;
            const int seg = (idx & 3) * 8;
            copy16_g2l(Bh + (long)(N0 + row) * K + k0 + seg, &b_s[buf][row][seg]);
        }
    };
    // stage B tile, !BT (f32 [K][N]): column-strided loads, packed b128 stores
    auto stageB_f32 = [&](int buf, int k0) {
        const int nb = tid & 127;
        const int kh = (tid >> 7) * 16;
        const float* src = Bf + (long)(k0 + kh) * N + N0 + nb;
        if (k0 + 32 < K) __builtin_prefetch(src + (long)32 * N, 0, 1);
        *(v8h*)&b_s[buf][nb][kh]     = pack8_strided(src, N);
        *(v8h*)&b_s[buf][nb][kh + 8] = pack8_strided(src + (long)8 * N, N);
    };
    auto stage = [&](int buf, int k0) {
        stageA(buf, k0);
        if (BT) stageB_half(buf, k0); else stageB_f32(buf, k0);
    };

    v8f acc00 = {}, acc01 = {}, acc10 = {}, acc11 = {};

    stage(0, 0);
    wait_async_all();
    __syncthreads();

    for (int k0 = 0; k0 < K; k0 += 32) {
        const int cur = (k0 >> 5) & 1;
        const int nxt = cur ^ 1;
        if (k0 + 32 < K) stage(nxt, k0 + 32);   // overlaps with WMMA below

        // A fragments: af[e] = A[m][ half*8 + (e&7) + (e>>3)*16 ]
        v16h af0 = join16(*(const v8h*)&a_s[cur][moff + r][half * 8],
                          *(const v8h*)&a_s[cur][moff + r][16 + half * 8]);
        v16h af1 = join16(*(const v8h*)&a_s[cur][moff + 16 + r][half * 8],
                          *(const v8h*)&a_s[cur][moff + 16 + r][16 + half * 8]);
        // B fragments: bf[e] = B[ half*8 + (e&7) + (e>>3)*16 ][ col ]
        v16h bf0 = join16(*(const v8h*)&b_s[cur][noff + r][half * 8],
                          *(const v8h*)&b_s[cur][noff + r][16 + half * 8]);
        v16h bf1 = join16(*(const v8h*)&b_s[cur][noff + 16 + r][half * 8],
                          *(const v8h*)&b_s[cur][noff + 16 + r][16 + half * 8]);

        acc00 = __builtin_amdgcn_wmma_f32_16x16x32_f16(
            false, af0, false, bf0, (short)0, acc00, false, false);
        acc01 = __builtin_amdgcn_wmma_f32_16x16x32_f16(
            false, af0, false, bf1, (short)0, acc01, false, false);
        acc10 = __builtin_amdgcn_wmma_f32_16x16x32_f16(
            false, af1, false, bf0, (short)0, acc10, false, false);
        acc11 = __builtin_amdgcn_wmma_f32_16x16x32_f16(
            false, af1, false, bf1, (short)0, acc11, false, false);

        wait_async_all();
        __syncthreads();
    }

    // epilogue: D layout -> m = v + half*8, col = lane&15
    const float*    zb = (MODE == 2) ? (z + (long)bz * N) : nullptr;
    const _Float16* vb = (MODE == 2) ? (vres + (long)bz * strideC) : nullptr;
    const int n0 = N0 + noff + r;
    const int n1 = n0 + 16;
    #pragma unroll
    for (int vv = 0; vv < 8; ++vv) {
        const int m0 = M0 + moff + half * 8 + vv;
        const int m1 = m0 + 16;
        if (MODE == 0) {
            const float s0 = scale[m0], b0 = bias[m0];
            const float s1 = scale[m1], b1 = bias[m1];
            Cb[(long)m0 * N + n0] = (OutT)(acc00[vv] * s0 + b0);
            Cb[(long)m0 * N + n1] = (OutT)(acc01[vv] * s0 + b0);
            Cb[(long)m1 * N + n0] = (OutT)(acc10[vv] * s1 + b1);
            Cb[(long)m1 * N + n1] = (OutT)(acc11[vv] * s1 + b1);
        } else if (MODE == 1) {
            Cb[(long)m0 * N + n0] = (OutT)(acc00[vv] * alpha);
            Cb[(long)m0 * N + n1] = (OutT)(acc01[vv] * alpha);
            Cb[(long)m1 * N + n0] = (OutT)(acc10[vv] * alpha);
            Cb[(long)m1 * N + n1] = (OutT)(acc11[vv] * alpha);
        } else {
            const float g0 = scale[m0], t0 = bias[m0];
            const float g1 = scale[m1], t1 = bias[m1];
            const float iz0 = 1.0f / zb[n0], iz1 = 1.0f / zb[n1];
            Cb[(long)m0 * N + n0] = (OutT)(acc00[vv] * iz0 * g0 + t0 + (float)vb[(long)m0 * N + n0]);
            Cb[(long)m0 * N + n1] = (OutT)(acc01[vv] * iz1 * g0 + t0 + (float)vb[(long)m0 * N + n1]);
            Cb[(long)m1 * N + n0] = (OutT)(acc10[vv] * iz0 * g1 + t1 + (float)vb[(long)m1 * N + n0]);
            Cb[(long)m1 * N + n1] = (OutT)(acc11[vv] * iz1 * g1 + t1 + (float)vb[(long)m1 * N + n1]);
        }
    }
}

// ---------------------------------------------------------------------------
// elementwise f32 -> f16 (weight conversion)
// ---------------------------------------------------------------------------
__global__ __launch_bounds__(256) void f32_to_f16_kernel(
    const float* __restrict__ src, _Float16* __restrict__ dst, int n)
{
    const int i = blockIdx.x * 256 + threadIdx.x;
    if (i < n) dst[i] = (_Float16)src[i];
}

// ---------------------------------------------------------------------------
// 5x5 depthwise conv + BN, then 1x1 depthwise (scale) + BN -> agg
// ---------------------------------------------------------------------------
__global__ __launch_bounds__(256) void dwconv_kernel(
    const float* __restrict__ in, const float* __restrict__ wdw,
    const float* __restrict__ sdw, const float* __restrict__ bdw,
    const float* __restrict__ wpw, const float* __restrict__ spw,
    const float* __restrict__ bpw, float* __restrict__ out)
{
    const int idx = blockIdx.x * 256 + threadIdx.x;
    const int n  = idx & (NPIX - 1);
    const int bc = idx >> 10;
    const int c  = bc % ATTN;
    const int h  = n >> 5;
    const int w  = n & 31;
    const float* src = in + (long)bc * NPIX;
    const float* wk  = wdw + c * 25;
    float acc = 0.f;
    #pragma unroll
    for (int dy = 0; dy < 5; ++dy) {
        const int yy = h + dy - 2;
        if (yy < 0 || yy > 31) continue;
        #pragma unroll
        for (int dx = 0; dx < 5; ++dx) {
            const int xx = w + dx - 2;
            if (xx < 0 || xx > 31) continue;
            acc += src[yy * 32 + xx] * wk[dy * 5 + dx];
        }
    }
    float t = acc * sdw[c] + bdw[c];
    out[idx] = t * (wpw[c] * spw[c]) + bpw[c];
}

// ---------------------------------------------------------------------------
// Assemble q (f32), k (f16), v (f16) from concat([qkv0, agg]) split.
// ---------------------------------------------------------------------------
__global__ __launch_bounds__(256) void assemble_kernel(
    const float* __restrict__ qkv0, const float* __restrict__ agg,
    float* __restrict__ q, _Float16* __restrict__ kh, _Float16* __restrict__ vh)
{
    const int idx = blockIdx.x * 256 + threadIdx.x;
    const int n  = idx & (NPIX - 1);
    const int bc = idx >> 10;
    const int c  = bc & (INNER - 1);
    const int b  = bc >> 9;
    const long base = (long)b * ATTN * NPIX;

    float qv = qkv0[base + (long)c * NPIX + n];
    q[idx] = fmaxf(qv, 0.f) + EPSV;

    float kv = (c < DHALF) ? qkv0[base + (long)(DIM + c) * NPIX + n]
                           : agg[base + (long)(c - DHALF) * NPIX + n];
    kh[idx] = (_Float16)(fmaxf(kv, 0.f) + EPSV);

    vh[idx] = (_Float16)agg[base + (long)(DHALF + c) * NPIX + n];
}

// ---------------------------------------------------------------------------
// kmean[b,c] = mean_n k[b,c,n]   (one wave32 per (b,c), shuffle reduction)
// ---------------------------------------------------------------------------
__global__ __launch_bounds__(256) void kmean_kernel(
    const _Float16* __restrict__ kh, float* __restrict__ kmean)
{
    const int lane = threadIdx.x & 31;
    const int wave = threadIdx.x >> 5;
    const int gw   = blockIdx.x * 8 + wave;          // (b*512 + c)
    const _Float16* kp = kh + (long)gw * NPIX;
    float s = 0.f;
    #pragma unroll 8
    for (int i = lane; i < NPIX; i += 32) s += (float)kp[i];
    #pragma unroll
    for (int off = 16; off > 0; off >>= 1) s += __shfl_down(s, off, 32);
    if (lane == 0) kmean[gw] = s * (1.0f / (float)NPIX);
}

// ---------------------------------------------------------------------------
// z[b,n] = sum_c kmean[b,c] * q[b,c,n] + eps
// ---------------------------------------------------------------------------
__global__ __launch_bounds__(256) void zrow_kernel(
    const float* __restrict__ q, const float* __restrict__ kmean,
    float* __restrict__ z)
{
    const int idx = blockIdx.x * 256 + threadIdx.x;   // b*NPIX + n
    const int n = idx & (NPIX - 1);
    const int b = idx >> 10;
    const float* km = kmean + b * INNER;
    const float* qb = q + (long)b * INNER * NPIX + n;
    float s = 0.f;
    #pragma unroll 4
    for (int c = 0; c < INNER; ++c) s += km[c] * qb[(long)c * NPIX];
    z[idx] = s + EPSV;
}

// ---------------------------------------------------------------------------
extern "C" void kernel_launch(void* const* d_in, const int* in_sizes, int n_in,
                              void* d_out, int out_size, void* d_ws, size_t ws_size,
                              hipStream_t stream) {
    const float* x      = (const float*)d_in[0];
    const float* w_qkv  = (const float*)d_in[1];
    const float* s_qkv  = (const float*)d_in[2];
    const float* b_qkv  = (const float*)d_in[3];
    const float* w_dw   = (const float*)d_in[4];
    const float* s_dw   = (const float*)d_in[5];
    const float* b_dw   = (const float*)d_in[6];
    const float* w_pw   = (const float*)d_in[7];
    const float* s_pw   = (const float*)d_in[8];
    const float* b_pw   = (const float*)d_in[9];
    const float* g_norm = (const float*)d_in[10];
    const float* bt_nrm = (const float*)d_in[11];
    const float* w_proj = (const float*)d_in[12];
    const float* s_proj = (const float*)d_in[13];
    const float* b_proj = (const float*)d_in[14];
    float* out = (float*)d_out;

    // workspace carve-up
    char* p = (char*)d_ws;
    float* qkv0 = (float*)p;             p += (long)BATCH * ATTN  * NPIX * 4;
    float* agg  = (float*)p;             p += (long)BATCH * ATTN  * NPIX * 4;
    float* q    = (float*)p;             p += (long)BATCH * INNER * NPIX * 4;
    float* y    = (float*)p;             p += (long)BATCH * INNER * NPIX * 4;
    _Float16* kh   = (_Float16*)p;       p += (long)BATCH * INNER * NPIX * 2;
    _Float16* vh   = (_Float16*)p;       p += (long)BATCH * INNER * NPIX * 2;
    _Float16* vkh  = (_Float16*)p;       p += (long)BATCH * INNER * INNER * 2;
    _Float16* wqh  = (_Float16*)p;       p += (long)ATTN * DIM * 2;
    _Float16* wph  = (_Float16*)p;       p += (long)DIM * INNER * 2;
    float* kmean = (float*)p;            p += (long)BATCH * INNER * 4;
    float* z     = (float*)p;            p += (long)BATCH * NPIX * 4;

    // 0) weights -> f16
    f32_to_f16_kernel<<<(ATTN * DIM + 255) / 256, 256, 0, stream>>>(w_qkv, wqh, ATTN * DIM);
    f32_to_f16_kernel<<<(DIM * INNER + 255) / 256, 256, 0, stream>>>(w_proj, wph, DIM * INNER);

    // 1) qkv0 = BN(W_qkv @ x): 768x1024, K=512 (weights shared across batch)
    gemm_wmma_kernel<0, false><<<dim3(NPIX / 128, ATTN / 64, BATCH), 256, 0, stream>>>(
        wqh, x, qkv0, ATTN, NPIX, DIM,
        0L, (long)DIM * NPIX, (long)ATTN * NPIX,
        s_qkv, b_qkv, 1.0f, nullptr, nullptr);

    // 2) agg = pw(BN(dw5x5(qkv0)))
    dwconv_kernel<<<(BATCH * ATTN * NPIX) / 256, 256, 0, stream>>>(
        qkv0, w_dw, s_dw, b_dw, w_pw, s_pw, b_pw, agg);

    // 3) split/concat -> q (f32), k (f16), v (f16)
    assemble_kernel<<<(BATCH * INNER * NPIX) / 256, 256, 0, stream>>>(
        qkv0, agg, q, kh, vh);

    // 4) kmean, z
    kmean_kernel<<<(BATCH * INNER) / 8, 256, 0, stream>>>(kh, kmean);
    zrow_kernel<<<(BATCH * NPIX) / 256, 256, 0, stream>>>(q, kmean, z);

    // 5) vk[b] = (1/N) * v[b] @ k[b]^T : 512x512, K=1024, f16 out, async B
    gemm_wmma_kernel<1, true><<<dim3(INNER / 128, INNER / 64, BATCH), 256, 0, stream>>>(
        vh, kh, vkh, INNER, INNER, NPIX,
        (long)INNER * NPIX, (long)INNER * NPIX, (long)INNER * INNER,
        nullptr, nullptr, 1.0f / (float)NPIX, nullptr, nullptr);

    // 6) y = (vk @ q)/z * g + bt + v : 512x1024, K=512
    gemm_wmma_kernel<2, false><<<dim3(NPIX / 128, INNER / 64, BATCH), 256, 0, stream>>>(
        vkh, q, y, INNER, NPIX, INNER,
        (long)INNER * INNER, (long)INNER * NPIX, (long)INNER * NPIX,
        g_norm, bt_nrm, 1.0f, z, vh);

    // 7) out = BN(W_proj @ y) : 512x1024, K=512 (weights shared)
    gemm_wmma_kernel<0, false><<<dim3(NPIX / 128, DIM / 64, BATCH), 256, 0, stream>>>(
        wph, y, out, DIM, NPIX, INNER,
        0L, (long)INNER * NPIX, (long)DIM * NPIX,
        s_proj, b_proj, 1.0f, nullptr, nullptr);
}
